// JointLoss_35605278883992
// MI455X (gfx1250) — matrix operand order repaired and checked by reading, not compiled, not run on hardware
//
#include <hip/hip_runtime.h>
#include <hip/hip_bf16.h>

// SimCLR / NT-Xent loss, fused:
//   loss = (1/2N) * sum_i [ log(sum_{j!=i} exp(2*s_ij)) - 2*s_{i, i^N} ]
// with s = cosine-sim of rows of representation [8192, 256].
// GEMM (8192x8192x256, 34.4 GFLOP) runs on the f16 WMMA pipe with fp32
// accumulation; the 4 MB normalized f16 matrix is L2-resident (192 MB L2),
// so the kernel is matrix-pipe bound, not HBM bound. Sim matrix never
// materialized; logits are in [-2,2] so LSE needs no max subtraction.

#define TWO_N 8192
#define NHALF 4096
#define DIM   256
#define INV_TAU 2.0f

typedef _Float16 v16h __attribute__((ext_vector_type(16)));
typedef _Float16 v8h  __attribute__((ext_vector_type(8)));
typedef float    v8f  __attribute__((ext_vector_type(8)));

// ---------------------------------------------------------------- kernel 1
// One block per row: fp32 sum-of-squares reduction, clamp by eps, store
// inv-norm and the f16 normalized row.
__global__ __launch_bounds__(256) void simclr_normalize(
    const float* __restrict__ rep, _Float16* __restrict__ x16,
    float* __restrict__ inv_norm) {
  __shared__ float sm[256];
  const int row = blockIdx.x;
  const int tid = threadIdx.x;
  const float v = rep[(size_t)row * DIM + tid];
  sm[tid] = v * v;
  __syncthreads();
  for (int s = 128; s > 0; s >>= 1) {
    if (tid < s) sm[tid] += sm[tid + s];
    __syncthreads();
  }
  const float inv = 1.0f / fmaxf(sqrtf(sm[0]), 1e-8f);
  if (tid == 0) inv_norm[row] = inv;
  x16[(size_t)row * DIM + tid] = (_Float16)(v * inv);
}

// ---------------------------------------------------------------- kernel 2
// Block = one 16-row tile (512 blocks), 8 waves/block (wave32).
// Each wave sweeps 64 of the 512 column tiles. Software-pipelined: while the
// 8x v_wmma_f32_16x16x32_f16 chain for tile ct runs, the 16 b128 loads for
// tile ct+8 are in flight. exp(2*s) fused into the epilogue; the main
// diagonal is masked only on the single tile that contains it.
//
// VGPR fragment layouts per CDNA5 ISA (CDNA5 ISA), khi = lane>>4:
//   A 16x32 f16 : lane L holds row M=L&15; K-halves {8*khi..8*khi+7} and
//                 {16+8*khi..16+8*khi+7}.
//   B 32x16 f16 : lane L holds col N=L&15; same K-halves as A.
//   C 16x16 f32 : VGPR v -> row v + 8*khi, col = L&15.
__global__ __launch_bounds__(256) void simclr_rowsum(
    const _Float16* __restrict__ x16, float* __restrict__ rowsum) {
  __shared__ float rowAcc[16];
  const int tid  = threadIdx.x;
  const int wid  = tid >> 5;
  const int lane = tid & 31;
  const int nIdx = lane & 15;
  const int khi  = lane >> 4;
  const int row0 = blockIdx.x * 16;
  const int NT   = TWO_N / 16;

  if (tid < 16) rowAcc[tid] = 0.0f;
  __syncthreads();

  // A fragments for this wave's 16-row tile, kept in registers (K=256 -> 8).
  union AV { v16h v; v8h h[2]; };
  AV a[8];
  const _Float16* Ar = x16 + (size_t)(row0 + nIdx) * DIM;
#pragma unroll
  for (int kk = 0; kk < 8; ++kk) {
    a[kk].h[0] = *(const v8h*)(Ar + kk * 32 + 8 * khi);
    a[kk].h[1] = *(const v8h*)(Ar + kk * 32 + 16 + 8 * khi);
  }

  float acc[8] = {0.f, 0.f, 0.f, 0.f, 0.f, 0.f, 0.f, 0.f};

  // Prime the pipeline: load B fragments for the first column tile.
  int ct = wid;
  v16h b[8];
  {
    const _Float16* Br = x16 + (size_t)(ct * 16 + nIdx) * DIM + 8 * khi;
#pragma unroll
    for (int kk = 0; kk < 8; ++kk) b[kk] = __builtin_shufflevector(*(const v8h*)(Br + kk * 32), *(const v8h*)(Br + kk * 32 + 16), 0, 1, 2, 3, 4, 5, 6, 7, 8, 9, 10, 11, 12, 13, 14, 15);
  }

  while (ct < NT) {
    const int col0 = ct * 16;
    const int ctn  = ct + 8;
    // Next tile's loads issue before the WMMA chain consumes this tile
    // (clamped on the last iteration so the address stays valid).
    const int ctl  = (ctn < NT) ? ctn : ct;
    const _Float16* Bn = x16 + (size_t)(ctl * 16 + nIdx) * DIM + 8 * khi;
    // Warm L2/L1 two tiles ahead (global_prefetch_b8).
    const int ctp = (ct + 16 < NT) ? ct + 16 : ct;
    __builtin_prefetch(x16 + (size_t)(ctp * 16 + nIdx) * DIM, 0, 0);
    v16h bn[8];
#pragma unroll
    for (int kk = 0; kk < 8; ++kk) bn[kk] = __builtin_shufflevector(*(const v8h*)(Bn + kk * 32), *(const v8h*)(Bn + kk * 32 + 16), 0, 1, 2, 3, 4, 5, 6, 7, 8, 9, 10, 11, 12, 13, 14, 15);

    v8f c = {};
#pragma unroll
    for (int kk = 0; kk < 8; ++kk) {
      c = __builtin_amdgcn_wmma_f32_16x16x32_f16(
          /*neg_a=*/false, a[kk].v, /*neg_b=*/false, b[kk],
          /*c_mod=*/(short)0, c, /*reuse_a=*/false, /*reuse_b=*/false);
    }

    float ex[8];
#pragma unroll
    for (int v = 0; v < 8; ++v) ex[v] = __expf(INV_TAU * c[v]);
    if (col0 == row0) {  // wave-uniform: only this tile holds the diagonal
#pragma unroll
      for (int v = 0; v < 8; ++v)
        if (nIdx == v + 8 * khi) ex[v] = 0.0f;
    }
#pragma unroll
    for (int v = 0; v < 8; ++v) acc[v] += ex[v];

#pragma unroll
    for (int kk = 0; kk < 8; ++kk) b[kk] = bn[kk];
    ct = ctn;
  }

  // Reduce each row's partial across the 16 lanes that share it, then merge
  // the 8 waves through LDS (ds_add_f32).
#pragma unroll
  for (int v = 0; v < 8; ++v) {
    float x = acc[v];
#pragma unroll
    for (int m = 8; m >= 1; m >>= 1) x += __shfl_xor(x, m, 32);
    if (nIdx == 0) atomicAdd(&rowAcc[v + 8 * khi], x);
  }
  __syncthreads();
  if (tid < 16) rowsum[row0 + tid] = rowAcc[tid];
}

// ---------------------------------------------------------------- kernel 3
// One wave per row: fp32 positive pair dot (partner = row ^ 4096), then
// per-block partial of log(rowsum) - 2*pos.  1024 blocks x 8 waves.
__global__ __launch_bounds__(256) void simclr_pos_partial(
    const float* __restrict__ rep, const float* __restrict__ inv_norm,
    const float* __restrict__ rowsum, float* __restrict__ partials) {
  __shared__ float wsum[8];
  const int tid  = threadIdx.x;
  const int wid  = tid >> 5;
  const int lane = tid & 31;
  const int r = blockIdx.x * 8 + wid;
  const int p = r ^ NHALF;  // (r + N) mod 2N, since 2N = 8192 = 2*4096
  const float* ar = rep + (size_t)r * DIM;
  const float* br = rep + (size_t)p * DIM;
  float s = 0.0f;
#pragma unroll
  for (int e = 0; e < 8; ++e) {
    const int idx = lane * 8 + e;
    s += ar[idx] * br[idx];
  }
#pragma unroll
  for (int m = 16; m >= 1; m >>= 1) s += __shfl_xor(s, m, 32);
  if (lane == 0) {
    const float pos = s * inv_norm[r] * inv_norm[p];
    wsum[wid] = logf(rowsum[r]) - INV_TAU * pos;
  }
  __syncthreads();
  if (tid == 0) {
    float t = 0.0f;
#pragma unroll
    for (int w = 0; w < 8; ++w) t += wsum[w];
    partials[blockIdx.x] = t;
  }
}

// ---------------------------------------------------------------- kernel 4
// Deterministic single-block reduction of the 1024 partials.
__global__ __launch_bounds__(256) void simclr_finalize(
    const float* __restrict__ partials, float* __restrict__ out) {
  __shared__ float sm[256];
  const int tid = threadIdx.x;
  sm[tid] = partials[tid] + partials[tid + 256] + partials[tid + 512] +
            partials[tid + 768];
  __syncthreads();
  for (int s = 128; s > 0; s >>= 1) {
    if (tid < s) sm[tid] += sm[tid + s];
    __syncthreads();
  }
  if (tid == 0) out[0] = sm[0] * (1.0f / (float)TWO_N);
}

// ---------------------------------------------------------------- launcher
extern "C" void kernel_launch(void* const* d_in, const int* in_sizes, int n_in,
                              void* d_out, int out_size, void* d_ws,
                              size_t ws_size, hipStream_t stream) {
  const float* rep = (const float*)d_in[0];
  float* out = (float*)d_out;

  // Workspace layout (~4.26 MB total):
  char* ws = (char*)d_ws;
  _Float16* x16     = (_Float16*)ws;                            // 4 MiB
  float*    invnorm = (float*)(ws + (size_t)4 * 1024 * 1024);   // 32 KiB
  float*    rowsum  = (float*)(ws + (size_t)4 * 1024 * 1024 + 32 * 1024);
  float*    parts   = (float*)(ws + (size_t)4 * 1024 * 1024 + 64 * 1024);

  simclr_normalize<<<TWO_N, 256, 0, stream>>>(rep, x16, invnorm);
  simclr_rowsum<<<TWO_N / 16, 256, 0, stream>>>(x16, rowsum);
  simclr_pos_partial<<<TWO_N / 8, 256, 0, stream>>>(rep, invnorm, rowsum, parts);
  simclr_finalize<<<1, 256, 0, stream>>>(parts, out);
}
